// AdaptiveLSTMCell_88115549045318
// MI455X (gfx1250) — compile-verified, matrix operand order checked by reading, not run on hardware
//
#include <hip/hip_runtime.h>
#include <hip/hip_bf16.h>
#include <math.h>

// Problem dims
#define B_DIM 4096
#define H_DIM 1024
#define TH_STRIDE 1152              // H + AH
#define OUT1 ((size_t)B_DIM * H_DIM)                     // start of new_total_h
#define OUT2 (OUT1 + (size_t)B_DIM * TH_STRIDE)          // start of new_total_c

typedef __attribute__((ext_vector_type(16))) __bf16 v16bf;
typedef __attribute__((ext_vector_type(2)))  __bf16 v2bf;
typedef __attribute__((ext_vector_type(8)))  float  v8f;
typedef __attribute__((ext_vector_type(4)))  float  f32x4;
typedef __attribute__((ext_vector_type(4)))  unsigned int u32x4;

union FragU { u32x4 q[2]; v16bf v; };

__device__ __forceinline__ unsigned pack2bf(float a, float b) {
#if __has_builtin(__builtin_amdgcn_cvt_pk_bf16_f32)
  union { v2bf v; unsigned u; } cv;
  cv.v = __builtin_amdgcn_cvt_pk_bf16_f32(a, b);
  return cv.u;
#else
  // round-half-up fp32->bf16, packed via one v_perm_b32:
  // result bytes = { ub[3], ub[2], ua[3], ua[2] }
  unsigned ua = __float_as_uint(a) + 0x8000u;
  unsigned ub = __float_as_uint(b) + 0x8000u;
  return __builtin_amdgcn_perm(ub, ua, 0x07060302u);
#endif
}

// Fragment gather per CDNA5 ISA 7.12.2 (16-bit A 16x32 / B 32x16 layout):
//   lane L: row = L&15 ; halves [kb .. kb+7] and [kb+16 .. kb+23], kb = (L&16)?8:0
// LDS tile row stride = 48 halves (96B, 16B-aligned) for b128 loads.
__device__ __forceinline__ v16bf ld_frag(const unsigned short* base, int lane) {
  int r  = lane & 15;
  int kb = (lane & 16) ? 8 : 0;
  FragU f;
  f.q[0] = *(const u32x4*)(base + r * 48 + kb);
  f.q[1] = *(const u32x4*)(base + r * 48 + kb + 16);
  return f.v;
}

__device__ __forceinline__ float sigm(float x) { return 1.0f / (1.0f + expf(-x)); }

__device__ __forceinline__ float wave_sum32(float v) {
  #pragma unroll
  for (int m = 16; m >= 1; m >>= 1) v += __shfl_xor(v, m, 32);
  return v;
}

// ---------------------------------------------------------------------------
// C[M,N] = A[M,K] @ W[N,K]^T  (fp32 in, bf16 WMMA, fp32 out; optional C += )
// Block tile 256(M) x 64(N), 256 threads = 8 waves.
// Wave w owns M-tiles {2w, 2w+1} x N-tiles {0..3}: 8 WMMAs / K-step.
// Requires M%256==0, N%64==0, K%32==0 (true for all call sites here).
// ---------------------------------------------------------------------------
__global__ __launch_bounds__(256) void gemm_bf16_wmma(
    const float* __restrict__ A, int lda,
    const float* __restrict__ W, int ldw,
    float* __restrict__ C, int ldc,
    int K, int accum)
{
  __shared__ __align__(16) unsigned short As[256 * 48];   // 24 KB
  __shared__ __align__(16) unsigned short Bs[64 * 48];    //  6 KB

  const int t    = threadIdx.x;
  const int wave = t >> 5;
  const int lane = t & 31;
  const int gmBase = blockIdx.y * 256;
  const int gnBase = blockIdx.x * 64;

  v8f acc[2][4];
  {
    v8f z = {0.f,0.f,0.f,0.f,0.f,0.f,0.f,0.f};
    #pragma unroll
    for (int m = 0; m < 2; ++m)
      #pragma unroll
      for (int j = 0; j < 4; ++j) acc[m][j] = z;
  }

  const int arow = t >> 1, aseg = (t & 1) * 16;   // A: 2x(128 rows x 32k), 32 floats/thread
  const int brow = t >> 2, bseg = (t & 3) * 8;    // B:  64 rows x 32k,  8 floats/thread
  const float* aPtr0 = A + (size_t)(gmBase + arow) * lda + aseg;
  const float* aPtr1 = aPtr0 + (size_t)128 * lda;
  const float* wPtr  = W + (size_t)(gnBase + brow) * ldw + bseg;

  for (int kk = 0; kk < K; kk += 32) {
    // ---- stage A tile rows [0,128) and [128,256) (fp32 -> bf16 -> LDS) ----
    {
      const f32x4* ap = (const f32x4*)(aPtr0 + kk);
      f32x4 f0 = ap[0], f1 = ap[1], f2 = ap[2], f3 = ap[3];
      u32x4 u0 = { pack2bf(f0.x, f0.y), pack2bf(f0.z, f0.w),
                   pack2bf(f1.x, f1.y), pack2bf(f1.z, f1.w) };
      u32x4 u1 = { pack2bf(f2.x, f2.y), pack2bf(f2.z, f2.w),
                   pack2bf(f3.x, f3.y), pack2bf(f3.z, f3.w) };
      u32x4* dst = (u32x4*)(As + arow * 48 + aseg);
      dst[0] = u0; dst[1] = u1;
    }
    {
      const f32x4* ap = (const f32x4*)(aPtr1 + kk);
      f32x4 f0 = ap[0], f1 = ap[1], f2 = ap[2], f3 = ap[3];
      u32x4 u0 = { pack2bf(f0.x, f0.y), pack2bf(f0.z, f0.w),
                   pack2bf(f1.x, f1.y), pack2bf(f1.z, f1.w) };
      u32x4 u1 = { pack2bf(f2.x, f2.y), pack2bf(f2.z, f2.w),
                   pack2bf(f3.x, f3.y), pack2bf(f3.z, f3.w) };
      u32x4* dst = (u32x4*)(As + (128 + arow) * 48 + aseg);
      dst[0] = u0; dst[1] = u1;
    }
    // ---- stage B tile ----
    {
      const f32x4* wp = (const f32x4*)(wPtr + kk);
      f32x4 f0 = wp[0], f1 = wp[1];
      u32x4 u = { pack2bf(f0.x, f0.y), pack2bf(f0.z, f0.w),
                  pack2bf(f1.x, f1.y), pack2bf(f1.z, f1.w) };
      *(u32x4*)(Bs + brow * 48 + bseg) = u;
    }
    if (kk + 32 < K) {
      __builtin_prefetch(aPtr0 + kk + 32, 0, 1);
      __builtin_prefetch(aPtr1 + kk + 32, 0, 1);
      __builtin_prefetch(wPtr + kk + 32, 0, 1);
    }
    __syncthreads();

    v16bf af0 = ld_frag(As + (2 * wave)     * 16 * 48, lane);
    v16bf af1 = ld_frag(As + (2 * wave + 1) * 16 * 48, lane);
    v16bf bf[4];
    #pragma unroll
    for (int j = 0; j < 4; ++j) bf[j] = ld_frag(Bs + j * 16 * 48, lane);
    #pragma unroll
    for (int j = 0; j < 4; ++j) {
      acc[0][j] = __builtin_amdgcn_wmma_f32_16x16x32_bf16(
                      false, af0, false, bf[j], (short)0, acc[0][j], false, false);
      acc[1][j] = __builtin_amdgcn_wmma_f32_16x16x32_bf16(
                      false, af1, false, bf[j], (short)0, acc[1][j], false, false);
    }
    __syncthreads();
  }

  // C/D layout: lanes 0-15 -> N=lane, M=v ; lanes 16-31 -> N=lane-16, M=8+v
  const int mo = (lane & 16) ? 8 : 0;
  const int no = lane & 15;
  #pragma unroll
  for (int m = 0; m < 2; ++m) {
    #pragma unroll
    for (int j = 0; j < 4; ++j) {
      const int gm = gmBase + (2 * wave + m) * 16 + mo;
      const int gn = gnBase + j * 16 + no;
      #pragma unroll
      for (int v = 0; v < 8; ++v) {
        float* cp = C + (size_t)(gm + v) * ldc + gn;
        float val = acc[m][j][v];
        if (accum) val += *cp;
        *cp = val;
      }
    }
  }
}

// ---------------------------------------------------------------------------
// Small-LSTM post: per row b: g = LN(ig)+LN(hg); acy = LN(cell); ada;
// fused zw/zb (dot-128 against (8*16,128) weights). Writes ada/acy output tails.
// ---------------------------------------------------------------------------
struct R4 { float x, y, z, w; };

__global__ __launch_bounds__(256) void small_post_kernel(
    const float* __restrict__ ig, const float* __restrict__ hg,
    const float* __restrict__ total_c,
    const float* __restrict__ a_ln_i_w, const float* __restrict__ a_ln_i_b,
    const float* __restrict__ a_ln_h_w, const float* __restrict__ a_ln_h_b,
    const float* __restrict__ a_ln_c_w, const float* __restrict__ a_ln_c_b,
    const float* __restrict__ norm_zw_w, const float* __restrict__ norm_zw_b,
    const float* __restrict__ bias_zb_w,
    float* __restrict__ ws_zw, float* __restrict__ ws_zb,
    float* __restrict__ out)
{
  __shared__ R4 red[256];
  __shared__ float gsh[512];
  __shared__ float cps[128];
  __shared__ float adash[128];

  const int b = blockIdx.x;
  const int t = threadIdx.x;

  const float i0 = ig[(size_t)b * 512 + t];
  const float i1 = ig[(size_t)b * 512 + 256 + t];
  const float h0 = hg[(size_t)b * 512 + t];
  const float h1 = hg[(size_t)b * 512 + 256 + t];

  R4 v; v.x = i0 + i1; v.y = i0*i0 + i1*i1; v.z = h0 + h1; v.w = h0*h0 + h1*h1;
  red[t] = v;
  __syncthreads();
  for (int s = 128; s > 0; s >>= 1) {
    if (t < s) {
      red[t].x += red[t+s].x; red[t].y += red[t+s].y;
      red[t].z += red[t+s].z; red[t].w += red[t+s].w;
    }
    __syncthreads();
  }
  R4 tot = red[0];
  const float mi  = tot.x * (1.0f/512.0f);
  const float rsi = rsqrtf(tot.y * (1.0f/512.0f) - mi*mi + 1e-5f);
  const float mh  = tot.z * (1.0f/512.0f);
  const float rsh = rsqrtf(tot.w * (1.0f/512.0f) - mh*mh + 1e-5f);

  gsh[t]       = (i0 - mi)*rsi*a_ln_i_w[t]     + a_ln_i_b[t]
               + (h0 - mh)*rsh*a_ln_h_w[t]     + a_ln_h_b[t];
  gsh[t + 256] = (i1 - mi)*rsi*a_ln_i_w[t+256] + a_ln_i_b[t+256]
               + (h1 - mh)*rsh*a_ln_h_w[t+256] + a_ln_h_b[t+256];
  __syncthreads();

  // adaptive cell: gi=[0:128) gf=[128:256) gj=[256:384) go=[384:512)
  float cp = 0.0f;
  if (t < 128) {
    const float gi = gsh[t], gf = gsh[128+t], gj = gsh[256+t];
    const float ac = total_c[(size_t)b * TH_STRIDE + 1024 + t];
    cp = sigm(gf)*ac + sigm(gi)*tanhf(gj);
    cps[t] = cp;
  }
  R4 v2; v2.x = (t < 128) ? cp : 0.f; v2.y = (t < 128) ? cp*cp : 0.f; v2.z = 0.f; v2.w = 0.f;
  red[t] = v2;
  __syncthreads();
  for (int s = 128; s > 0; s >>= 1) {
    if (t < s) { red[t].x += red[t+s].x; red[t].y += red[t+s].y; }
    __syncthreads();
  }
  const float mc  = red[0].x * (1.0f/128.0f);
  const float rsc = rsqrtf(red[0].y * (1.0f/128.0f) - mc*mc + 1e-5f);

  if (t < 128) {
    const float acy = (cps[t] - mc)*rsc*a_ln_c_w[t] + a_ln_c_b[t];
    const float go  = gsh[384 + t];
    const float ada = sigm(go) * tanhf(acy);
    adash[t] = ada;
    out[OUT1 + (size_t)b * TH_STRIDE + 1024 + t] = ada;   // new_total_h tail
    out[OUT2 + (size_t)b * TH_STRIDE + 1024 + t] = acy;   // new_total_c tail
  }
  __syncthreads();

  // zw[b,s,i] = ada . norm_zw_w[s,i,:] + norm_zw_b ; zb similar (no bias)
  {
    const int  idx  = (t < 128) ? t : (t - 128);
    const float* ws = (t < 128) ? (norm_zw_w + (size_t)idx * 128)
                                : (bias_zb_w + (size_t)idx * 128);
    float acc = 0.f;
    #pragma unroll 4
    for (int a = 0; a < 128; a += 4) {
      acc += adash[a]*ws[a] + adash[a+1]*ws[a+1]
           + adash[a+2]*ws[a+2] + adash[a+3]*ws[a+3];
    }
    if (t < 128) ws_zw[(size_t)b * 128 + idx] = acc + norm_zw_b[idx];
    else         ws_zb[(size_t)b * 128 + idx] = acc;
  }
}

// ---------------------------------------------------------------------------
// Main post: 4 rows/block, one wave32 per row. On-the-fly alpha/beta (K=16),
// two 4096-wide LNs, c-path LN, writes hy / new_total_h / new_total_c heads.
// ---------------------------------------------------------------------------
__global__ __launch_bounds__(128) void main_post_kernel(
    const float* __restrict__ xh, const float* __restrict__ hh,
    const float* __restrict__ zw, const float* __restrict__ zb,
    const float* __restrict__ norm_alpha_w, const float* __restrict__ bias_beta_w,
    const float* __restrict__ b_ih, const float* __restrict__ b_hh,
    const float* __restrict__ ln_i_w, const float* __restrict__ ln_i_b,
    const float* __restrict__ ln_h_w, const float* __restrict__ ln_h_b,
    const float* __restrict__ ln_c_w, const float* __restrict__ ln_c_b,
    const float* __restrict__ total_c,
    float* __restrict__ out)
{
  extern __shared__ float dyn[];              // 4*4096 (ing) + 4*4096 (hidg)
  __shared__ float zwzb[4][256];

  const int t    = threadIdx.x;
  const int wave = t >> 5;
  const int lane = t & 31;
  const int b    = blockIdx.x * 4 + wave;

  float* ing  = dyn + wave * 4096;
  float* hidg = dyn + 4 * 4096 + wave * 4096;

  for (int i = lane; i < 128; i += 32) {
    zwzb[wave][i]       = zw[(size_t)b * 128 + i];
    zwzb[wave][128 + i] = zb[(size_t)b * 128 + i];
  }
  // wave-private LDS slice; compiler inserts dscnt waits for same-wave RAW.

  // ---- in_g ----
  float s1 = 0.f, sq1 = 0.f;
  for (int j = lane; j < 4096; j += 32) {
    const int s = j >> 10, h = j & 1023;
    const f32x4* aw = (const f32x4*)(norm_alpha_w + (size_t)((s << 10) | h) * 16);
    const f32x4* bw = (const f32x4*)(bias_beta_w  + (size_t)((s << 10) | h) * 16);
    const float* zr = &zwzb[wave][s * 16];
    const float* br = &zwzb[wave][128 + s * 16];
    float alpha = 0.f, beta = 0.f;
    #pragma unroll
    for (int q = 0; q < 4; ++q) {
      f32x4 a4 = aw[q], b4 = bw[q];
      alpha += zr[4*q]*a4.x + zr[4*q+1]*a4.y + zr[4*q+2]*a4.z + zr[4*q+3]*a4.w;
      beta  += br[4*q]*b4.x + br[4*q+1]*b4.y + br[4*q+2]*b4.z + br[4*q+3]*b4.w;
    }
    const float val = xh[(size_t)b * 4096 + j] * alpha + b_ih[j] + beta;
    ing[j] = val; s1 += val; sq1 += val * val;
  }
  const float mi  = wave_sum32(s1)  * (1.0f/4096.0f);
  const float rsi = rsqrtf(wave_sum32(sq1) * (1.0f/4096.0f) - mi*mi + 1e-5f);

  // ---- hid_g (s in 4..7) ----
  float s2 = 0.f, sq2 = 0.f;
  for (int j = lane; j < 4096; j += 32) {
    const int s = 4 + (j >> 10), h = j & 1023;
    const f32x4* aw = (const f32x4*)(norm_alpha_w + (size_t)((s << 10) | h) * 16);
    const f32x4* bw = (const f32x4*)(bias_beta_w  + (size_t)((s << 10) | h) * 16);
    const float* zr = &zwzb[wave][s * 16];
    const float* br = &zwzb[wave][128 + s * 16];
    float alpha = 0.f, beta = 0.f;
    #pragma unroll
    for (int q = 0; q < 4; ++q) {
      f32x4 a4 = aw[q], b4 = bw[q];
      alpha += zr[4*q]*a4.x + zr[4*q+1]*a4.y + zr[4*q+2]*a4.z + zr[4*q+3]*a4.w;
      beta  += br[4*q]*b4.x + br[4*q+1]*b4.y + br[4*q+2]*b4.z + br[4*q+3]*b4.w;
    }
    const float val = hh[(size_t)b * 4096 + j] * alpha + b_hh[j] + beta;
    hidg[j] = val; s2 += val; sq2 += val * val;
  }
  const float mh  = wave_sum32(s2)  * (1.0f/4096.0f);
  const float rsh = rsqrtf(wave_sum32(sq2) * (1.0f/4096.0f) - mh*mh + 1e-5f);

  // gates = LN(in_g) + LN(hid_g)   (overwrite ing)
  for (int j = lane; j < 4096; j += 32) {
    ing[j] = (ing[j]  - mi)*rsi*ln_i_w[j] + ln_i_b[j]
           + (hidg[j] - mh)*rsh*ln_h_w[j] + ln_h_b[j];
  }

  // main cell: i=[0:1024) j=[1024:2048) f=[2048:3072) o=[3072:4096)
  float c1 = 0.f, c2 = 0.f;
  for (int h = lane; h < 1024; h += 32) {
    const float gi = ing[h], gj = ing[1024 + h], gf = ing[2048 + h];
    const float cx = total_c[(size_t)b * TH_STRIDE + h];
    const float cp = sigm(gf)*cx + sigm(gi)*tanhf(gj);
    hidg[h] = cp; c1 += cp; c2 += cp * cp;
  }
  const float mcc = wave_sum32(c1) * (1.0f/1024.0f);
  const float rsc = rsqrtf(wave_sum32(c2) * (1.0f/1024.0f) - mcc*mcc + 1e-5f);

  for (int h = lane; h < 1024; h += 32) {
    const float cy = (hidg[h] - mcc)*rsc*ln_c_w[h] + ln_c_b[h];
    const float go = ing[3072 + h];
    const float hy = sigm(go) * tanhf(cy);
    out[(size_t)b * H_DIM + h]              = hy;   // hy
    out[OUT1 + (size_t)b * TH_STRIDE + h]   = hy;   // new_total_h head
    out[OUT2 + (size_t)b * TH_STRIDE + h]   = cy;   // new_total_c head
  }
}

// ---------------------------------------------------------------------------
extern "C" void kernel_launch(void* const* d_in, const int* in_sizes, int n_in,
                              void* d_out, int out_size, void* d_ws, size_t ws_size,
                              hipStream_t stream) {
  const float* x           = (const float*)d_in[0];
  const float* total_h     = (const float*)d_in[1];
  const float* total_c     = (const float*)d_in[2];
  const float* w_ih        = (const float*)d_in[3];
  const float* w_hh        = (const float*)d_in[4];
  const float* b_ih        = (const float*)d_in[5];
  const float* b_hh        = (const float*)d_in[6];
  const float* ln_i_w      = (const float*)d_in[7];
  const float* ln_i_b      = (const float*)d_in[8];
  const float* ln_h_w      = (const float*)d_in[9];
  const float* ln_h_b      = (const float*)d_in[10];
  const float* ln_c_w      = (const float*)d_in[11];
  const float* ln_c_b      = (const float*)d_in[12];
  const float* norm_zw_w   = (const float*)d_in[13];
  const float* norm_zw_b   = (const float*)d_in[14];
  const float* norm_alpha_w= (const float*)d_in[15];
  const float* bias_zb_w   = (const float*)d_in[16];
  const float* bias_beta_w = (const float*)d_in[17];
  const float* a_w_ih      = (const float*)d_in[18];
  const float* a_w_hh      = (const float*)d_in[19];
  const float* a_ln_i_w    = (const float*)d_in[20];
  const float* a_ln_i_b    = (const float*)d_in[21];
  const float* a_ln_h_w    = (const float*)d_in[22];
  const float* a_ln_h_b    = (const float*)d_in[23];
  const float* a_ln_c_w    = (const float*)d_in[24];
  const float* a_ln_c_b    = (const float*)d_in[25];
  float* out = (float*)d_out;
  float* ws  = (float*)d_ws;

  // Workspace layout (floats): ~148 MB total
  float* XH = ws;                                  // 4096*4096
  float* HH = XH + (size_t)4096 * 4096;            // 4096*4096
  float* IG = HH + (size_t)4096 * 4096;            // 4096*512
  float* HG = IG + (size_t)4096 * 512;             // 4096*512
  float* ZW = HG + (size_t)4096 * 512;             // 4096*128
  float* ZB = ZW + (size_t)4096 * 128;             // 4096*128

  dim3 blk(256);
  // xh = x @ w_ih^T                       (M=4096, N=4096, K=1024)
  gemm_bf16_wmma<<<dim3(64, 16), blk, 0, stream>>>(x, 1024, w_ih, 1024, XH, 4096, 1024, 0);
  // hh = hx @ w_hh^T   (hx = total_h[:, :1024], row stride 1152)
  gemm_bf16_wmma<<<dim3(64, 16), blk, 0, stream>>>(total_h, TH_STRIDE, w_hh, 1024, HH, 4096, 1024, 0);
  // ig = [x | hx] @ a_w_ih^T  (split-K over the concat)
  gemm_bf16_wmma<<<dim3(8, 16), blk, 0, stream>>>(x, 1024, a_w_ih, 2048, IG, 512, 1024, 0);
  gemm_bf16_wmma<<<dim3(8, 16), blk, 0, stream>>>(total_h, TH_STRIDE, a_w_ih + 1024, 2048, IG, 512, 1024, 1);
  // hg = ah @ a_w_hh^T  (ah = total_h[:, 1024:], K=128)
  gemm_bf16_wmma<<<dim3(8, 16), blk, 0, stream>>>(total_h + 1024, TH_STRIDE, a_w_hh, 128, HG, 512, 128, 0);

  small_post_kernel<<<4096, 256, 0, stream>>>(
      IG, HG, total_c, a_ln_i_w, a_ln_i_b, a_ln_h_w, a_ln_h_b, a_ln_c_w, a_ln_c_b,
      norm_zw_w, norm_zw_b, bias_zb_w, ZW, ZB, out);

  main_post_kernel<<<1024, 128, (size_t)8 * 4096 * sizeof(float), stream>>>(
      XH, HH, ZW, ZB, norm_alpha_w, bias_beta_w, b_ih, b_hh,
      ln_i_w, ln_i_b, ln_h_w, ln_h_b, ln_c_w, ln_c_b, total_c, out);
}
